// GC_WeightNet_7559142441145
// MI455X (gfx1250) — matrix operand order repaired and checked by reading, not compiled, not run on hardware
//
#include <hip/hip_runtime.h>
#include <hip/hip_bf16.h>

// ---------------- problem constants (match reference) ----------------
#define BATCH    16
#define CIN      128
#define COUT     128
#define HH       56
#define WW       56
#define SPATIAL  (HH * WW)          // 3136
#define KTOT     (CIN * 9)          // 1152 (implicit-GEMM K)
#define RDIM     16                 // bottleneck width
#define MOUT     256                // M * C_OUT
#define OUTPG    576                // COUT_FC / GROUPS
#define COUTFC   147456             // 128*128*3*3
#define LN_EPS   1e-5f

typedef __attribute__((ext_vector_type(16))) _Float16 v16h;
typedef __attribute__((ext_vector_type(8)))  float    v8f;

// =====================================================================
// Kernel 1: GC attention + bottleneck MLP.  One block (256 thr) / sample.
// =====================================================================
__global__ __launch_bounds__(256) void k_attention(
    const float* __restrict__ x,
    const float* __restrict__ w_mask, const float* __restrict__ b_mask,
    const float* __restrict__ w1, const float* __restrict__ b1,
    const float* __restrict__ ln_g, const float* __restrict__ ln_b,
    const float* __restrict__ w2, const float* __restrict__ b2,
    float* __restrict__ a_out)
{
    __shared__ __attribute__((aligned(16))) float sm[SPATIAL];   // mask -> att
    __shared__ float sred[256];
    __shared__ float sctx[CIN];
    __shared__ float st[RDIM];

    const int b   = blockIdx.x;
    const int tid = threadIdx.x;
    const float* xb = x + (size_t)b * CIN * SPATIAL;

    // ---- mask[s] = sum_c x[c,s] * w_mask[c] + b_mask ----
    for (int s = tid; s < SPATIAL; s += 256) {
        float acc = b_mask[0];
        #pragma unroll 4
        for (int c = 0; c < CIN; ++c)
            acc += xb[(size_t)c * SPATIAL + s] * w_mask[c];
        sm[s] = acc;
    }
    __syncthreads();

    // ---- softmax over SPATIAL ----
    float mx = -3.0e38f;
    for (int s = tid; s < SPATIAL; s += 256) mx = fmaxf(mx, sm[s]);
    sred[tid] = mx;
    __syncthreads();
    for (int off = 128; off > 0; off >>= 1) {
        if (tid < off) sred[tid] = fmaxf(sred[tid], sred[tid + off]);
        __syncthreads();
    }
    mx = sred[0];
    __syncthreads();

    float sum = 0.f;
    for (int s = tid; s < SPATIAL; s += 256) {
        float e = __expf(sm[s] - mx);
        sm[s] = e;
        sum  += e;
    }
    sred[tid] = sum;
    __syncthreads();
    for (int off = 128; off > 0; off >>= 1) {
        if (tid < off) sred[tid] += sred[tid + off];
        __syncthreads();
    }
    const float inv = 1.0f / sred[0];
    __syncthreads();

    // ---- ctx[c] = sum_s x[c,s] * att[s] ----
    {
        const int c = tid & (CIN - 1);
        const int h = tid >> 7;
        const int s0 = h * (SPATIAL / 2), s1 = s0 + (SPATIAL / 2);
        float acc = 0.f;
        for (int s = s0; s < s1; ++s)
            acc += xb[(size_t)c * SPATIAL + s] * sm[s];
        sred[tid] = acc * inv;
        __syncthreads();
        if (h == 0) sctx[c] = sred[c] + sred[c + 128];
    }
    __syncthreads();

    // ---- t = ctx @ w1.T + b1  (R = 16) ----
    if (tid < RDIM) {
        float acc = b1[tid];
        for (int c = 0; c < CIN; ++c) acc += sctx[c] * w1[tid * CIN + c];
        st[tid] = acc;
    }
    __syncthreads();

    // ---- LayerNorm(16) + ReLU ----
    if (tid == 0) {
        float mu = 0.f;
        for (int r = 0; r < RDIM; ++r) mu += st[r];
        mu *= (1.0f / RDIM);
        float var = 0.f;
        for (int r = 0; r < RDIM; ++r) { float d = st[r] - mu; var += d * d; }
        var *= (1.0f / RDIM);
        const float rstd = rsqrtf(var + LN_EPS);
        for (int r = 0; r < RDIM; ++r) {
            float v = (st[r] - mu) * rstd * ln_g[r] + ln_b[r];
            st[r] = fmaxf(v, 0.f);
        }
    }
    __syncthreads();

    // ---- a = sigmoid(t @ w2.T + b2) ----
    {
        float acc = b2[tid];
        #pragma unroll
        for (int r = 0; r < RDIM; ++r) acc += st[r] * w2[tid * RDIM + r];
        a_out[b * MOUT + tid] = 1.0f / (1.0f + __expf(-acc));
    }
}

// =====================================================================
// Kernel 2: generated per-sample conv weights -> f16 A-matrix, K
// reordered as kappa = (kh*3+kw)*128 + ci.
// =====================================================================
__global__ __launch_bounds__(256) void k_weightgen(
    const float* __restrict__ a, const float* __restrict__ w_fc,
    const float* __restrict__ b_fc, _Float16* __restrict__ Wh)
{
    const size_t i = (size_t)blockIdx.x * 256 + threadIdx.x;
    if (i >= (size_t)BATCH * COUTFC) return;
    const int b = (int)(i / COUTFC);
    const int f = (int)(i - (size_t)b * COUTFC);
    const float v = a[b * MOUT + f / OUTPG] * w_fc[f] + b_fc[f];

    const int co  = f / (CIN * 9);
    const int rem = f - co * (CIN * 9);
    const int ci  = rem / 9;
    const int r   = rem - ci * 9;          // kh*3 + kw
    Wh[(size_t)b * COUTFC + (size_t)co * KTOT + r * CIN + ci] = (_Float16)v;
}

// =====================================================================
// Kernel 3: x -> f16 (B-matrix source)
// =====================================================================
__global__ __launch_bounds__(256) void k_x2h(
    const float* __restrict__ x, _Float16* __restrict__ Xh)
{
    const size_t i = (size_t)blockIdx.x * 256 + threadIdx.x;
    if (i >= (size_t)BATCH * CIN * SPATIAL) return;
    Xh[i] = (_Float16)x[i];
}

// =====================================================================
// Kernel 4: per-sample implicit-GEMM conv via WMMA f16 (f32 accum)
//   grid = (3136/64, 16), block = 256 (8 waves).
//   A-tile staged with GLOBAL_LOAD_ASYNC_TO_LDS_B128 (ASYNCcnt path).
// =====================================================================
#define NT   64     // spatial tile
#define LDA  40     // padded LDS row stride (halfs) for A (128 x 32)
#define LDB  40     // padded LDS row stride (halfs) for B (64 x 32, n-major)

__device__ inline v16h ld_frag(const _Float16* p0, const _Float16* p1)
{
    union { v16h v; float4 f[2]; } u;
    u.f[0] = *(const float4*)p0;
    u.f[1] = *(const float4*)p1;
    return u.v;
}

__global__ __launch_bounds__(256) void k_conv(
    const _Float16* __restrict__ Xh,
    const _Float16* __restrict__ Wh,
    float* __restrict__ out)
{
    __shared__ __attribute__((aligned(16))) _Float16 sA[COUT * LDA]; // 10.0 KB
    __shared__ __attribute__((aligned(16))) _Float16 sB[NT * LDB];   //  5.0 KB

    const int b      = blockIdx.y;
    const int s_base = blockIdx.x * NT;
    const int tid    = threadIdx.x;
    const int lane   = tid & 31;
    const int wave   = tid >> 5;

    const _Float16* Wb = Wh + (size_t)b * COUT * KTOT;
    const _Float16* Xb = Xh + (size_t)b * CIN * SPATIAL;

    // --- A staging (async global->LDS): 2 x 16B chunks per thread ---
    // chunk c in {tid, tid+256}: row = c>>2 (0..127), off8 = (c&3)*8 halfs
    const int ar0 = tid >> 2,          ao0 = (tid & 3) * 8;
    const int ar1 = (tid + 256) >> 2,  ao1 = (tid & 3) * 8;
    const unsigned la0 = (unsigned)(size_t)&sA[ar0 * LDA + ao0];  // LDS byte offset
    const unsigned la1 = (unsigned)(size_t)&sA[ar1 * LDA + ao1];
    const _Float16* ga0base = Wb + (size_t)ar0 * KTOT + ao0;
    const _Float16* ga1base = Wb + (size_t)ar1 * KTOT + ao1;

    // per-thread B-staging coordinates (K-loop-invariant):
    // element e = tid + j*256  ->  kk = (tid>>6) + 4j,  nn = tid & 63
    const int nn  = tid & 63;
    const int kk0 = tid >> 6;                  // 0..3
    const int sN  = s_base + nn;
    const int y0  = sN / WW;
    const int x0  = sN - y0 * WW;
    _Float16* sBp = &sB[nn * LDB + kk0];

    // A fragment LDS coords (contiguous k = 16*(lane>>4) + s, same as B)
    const int arow = (wave << 4) + (lane & 15);
    const int akc  = (lane >> 4) * 16;
    // B fragment LDS coords
    const int bn = lane & 15;
    const int bk = (lane >> 4) * 16;

    v8f acc0 = {}, acc1 = {}, acc2 = {}, acc3 = {};

    for (int k0 = 0; k0 < KTOT; k0 += 32) {
        // ---- per-iteration scalar K decomposition: k = off*128 + ci ----
        const int off    = k0 >> 7;            // 0..8  == kh*3+kw
        const int kh     = off / 3;
        const int kw     = off - kh * 3;
        const int cibase = k0 - (off << 7);    // 0,32,64,96

        // ---- stage A slice via async LDS loads (no VGPR round-trip) ----
        {
            const unsigned long long g0 = (unsigned long long)(ga0base + k0);
            const unsigned long long g1 = (unsigned long long)(ga1base + k0);
            asm volatile("global_load_async_to_lds_b128 %0, %1, off"
                         :: "v"(la0), "v"(g0) : "memory");
            asm volatile("global_load_async_to_lds_b128 %0, %1, off"
                         :: "v"(la1), "v"(g1) : "memory");
        }
        if (k0 + 32 < KTOT)
            __builtin_prefetch(ga0base + k0 + 32, 0, 3);

        // ---- stage B slice: im2col 32 (K) x 64 (N), branch-free ----
        {
            const int yy = y0 + kh - 1;
            const int xw = x0 + kw - 1;
            const bool inb = ((unsigned)yy < (unsigned)HH) &
                             ((unsigned)xw < (unsigned)WW);
            const int yc = yy < 0 ? 0 : (yy > HH - 1 ? HH - 1 : yy);
            const int xc = xw < 0 ? 0 : (xw > WW - 1 ? WW - 1 : xw);
            const _Float16* src =
                Xb + (size_t)(cibase + kk0) * SPATIAL + yc * WW + xc;
            #pragma unroll
            for (int j = 0; j < 8; ++j) {
                const _Float16 v = src[(size_t)(4 * j) * SPATIAL];
                sBp[4 * j] = inb ? v : (_Float16)0.f;
            }
        }

        // async A-tile must land before anyone reads sA
        asm volatile("s_wait_asynccnt 0x0" ::: "memory");
        __syncthreads();

        // ---- fragments + 4 WMMAs per wave ----
        const v16h afr = ld_frag(&sA[arow * LDA + akc],
                                 &sA[arow * LDA + akc + 8]);
        const v16h b0 = ld_frag(&sB[(bn     ) * LDB + bk], &sB[(bn     ) * LDB + bk + 8]);
        const v16h b1 = ld_frag(&sB[(bn + 16) * LDB + bk], &sB[(bn + 16) * LDB + bk + 8]);
        const v16h b2 = ld_frag(&sB[(bn + 32) * LDB + bk], &sB[(bn + 32) * LDB + bk + 8]);
        const v16h b3 = ld_frag(&sB[(bn + 48) * LDB + bk], &sB[(bn + 48) * LDB + bk + 8]);

        acc0 = __builtin_amdgcn_wmma_f32_16x16x32_f16(false, afr, false, b0, (short)0, acc0, false, false);
        acc1 = __builtin_amdgcn_wmma_f32_16x16x32_f16(false, afr, false, b1, (short)0, acc1, false, false);
        acc2 = __builtin_amdgcn_wmma_f32_16x16x32_f16(false, afr, false, b2, (short)0, acc2, false, false);
        acc3 = __builtin_amdgcn_wmma_f32_16x16x32_f16(false, afr, false, b3, (short)0, acc3, false, false);

        __syncthreads();
    }

    // ---- store D: lane N = lane&15, rows M = v + 8*(lane>>4) ----
    const int ncol  = lane & 15;
    const int mrow0 = (lane >> 4) * 8;
    float* ob = out + ((size_t)b * COUT + (wave << 4)) * SPATIAL + s_base;
    #pragma unroll
    for (int v = 0; v < 8; ++v) {
        float* row = ob + (size_t)(mrow0 + v) * SPATIAL + ncol;
        row[ 0] = acc0[v];
        row[16] = acc1[v];
        row[32] = acc2[v];
        row[48] = acc3[v];
    }
}

// =====================================================================
// launch
// =====================================================================
extern "C" void kernel_launch(void* const* d_in, const int* in_sizes, int n_in,
                              void* d_out, int out_size, void* d_ws, size_t ws_size,
                              hipStream_t stream)
{
    const float* x      = (const float*)d_in[0];
    const float* w_mask = (const float*)d_in[1];
    const float* b_mask = (const float*)d_in[2];
    const float* w1     = (const float*)d_in[3];
    const float* b1     = (const float*)d_in[4];
    const float* ln_g   = (const float*)d_in[5];
    const float* ln_b   = (const float*)d_in[6];
    const float* w2     = (const float*)d_in[7];
    const float* b2     = (const float*)d_in[8];
    const float* w_fc   = (const float*)d_in[9];
    const float* b_fc   = (const float*)d_in[10];
    float* out = (float*)d_out;

    // workspace layout (16B aligned slices)
    char* ws = (char*)d_ws;
    float*    a_buf = (float*)ws;                                   // 16 KB
    _Float16* Wh    = (_Float16*)(ws + 16384);                      // 4.5 MB
    _Float16* Xh    = (_Float16*)(ws + 16384 + (size_t)BATCH * COUTFC * 2);

    k_attention<<<BATCH, 256, 0, stream>>>(x, w_mask, b_mask, w1, b1,
                                           ln_g, ln_b, w2, b2, a_buf);

    k_weightgen<<<(BATCH * COUTFC) / 256, 256, 0, stream>>>(a_buf, w_fc, b_fc, Wh);

    k_x2h<<<(BATCH * CIN * SPATIAL) / 256, 256, 0, stream>>>(x, Xh);

    dim3 grid(SPATIAL / NT, BATCH);   // (49, 16)
    k_conv<<<grid, 256, 0, stream>>>(Xh, Wh, out);
}
